// GCNmfConv_2688649527504
// MI455X (gfx1250) — matrix operand order repaired and checked by reading, not compiled, not run on hardware
//
#include <hip/hip_runtime.h>

#define NN      50000       // nodes
#define FF      256         // F_IN == F_OUT
#define KC      5           // mixture components
#define EE      1600000     // edges
#define MTILES  3125        // NN / 16

typedef __attribute__((ext_vector_type(16))) __bf16        v16bf;
typedef __attribute__((ext_vector_type(8)))  float         v8f;
typedef __attribute__((ext_vector_type(4)))  unsigned int  v4u;

__device__ __forceinline__ unsigned short f2bf(float f) {
    unsigned u = __float_as_uint(f);
    u += 0x7FFFu + ((u >> 16) & 1u);          // round-to-nearest-even
    return (unsigned short)(u >> 16);
}

// ---------------------------------------------------------------------------
// prep: Wt[o][f] = bf16(W[f][o]);  Wsq[o][f] = bf16(W[f][o]^2);  var = exp(lv)
// grid: 256 + KC blocks x 256 threads
// ---------------------------------------------------------------------------
__global__ void prep_kernel(const float* __restrict__ weight,
                            const float* __restrict__ logvars,
                            unsigned short* __restrict__ wt,
                            unsigned short* __restrict__ wsq,
                            float* __restrict__ var) {
    int b = blockIdx.x, t = threadIdx.x;
    if (b < FF) {
        float w = weight[b * FF + t];          // W[f=b][o=t]
        wt [t * FF + b] = f2bf(w);
        wsq[t * FF + b] = f2bf(w * w);
    } else {
        int k = b - FF;
        if (k < KC) var[k * FF + t] = expf(logvars[k * FF + t]);
    }
}

// ---------------------------------------------------------------------------
// gamma: softmax_k( logp[k] - 0.5 * sum_{f:!mask} (x-mean)^2/var )
// (constants across k cancel in the softmax)
// one wave32 per node; grid 6250 x 256
// ---------------------------------------------------------------------------
__global__ void gamma_kernel(const float* __restrict__ x,
                             const unsigned char* __restrict__ mask,
                             const float* __restrict__ logp,
                             const float* __restrict__ means,
                             const float* __restrict__ var,
                             float* __restrict__ gamma) {
    int wave = threadIdx.x >> 5;
    int lane = threadIdx.x & 31;
    int n = blockIdx.x * 8 + wave;
    if (n >= NN) return;

    float p[KC] = {0.f, 0.f, 0.f, 0.f, 0.f};
    for (int f = lane; f < FF; f += 32) {
        if (!mask[(size_t)n * FF + f]) {
            float xv = x[(size_t)n * FF + f];
            #pragma unroll
            for (int k = 0; k < KC; ++k) {
                float d = xv - means[k * FF + f];
                p[k] += d * d / var[k * FF + f];
            }
        }
    }
    #pragma unroll
    for (int k = 0; k < KC; ++k)
        for (int off = 16; off > 0; off >>= 1)
            p[k] += __shfl_xor(p[k], off, 32);

    float g[KC], mx = -1e30f;
    #pragma unroll
    for (int k = 0; k < KC; ++k) { g[k] = logp[k] - 0.5f * p[k]; mx = fmaxf(mx, g[k]); }
    float s = 0.f;
    #pragma unroll
    for (int k = 0; k < KC; ++k) { g[k] = expf(g[k] - mx); s += g[k]; }
    float inv = 1.f / s;
    if (lane == 0) {
        #pragma unroll
        for (int k = 0; k < KC; ++k) gamma[(size_t)k * NN + n] = g[k] * inv;
    }
}

// ---------------------------------------------------------------------------
// gemm: per block = one (k, 16-node strip); 16 waves x one 16x16 output tile.
// A (mean/var, masked select) staged in LDS as bf16; 8 k-steps of
// v_wmma_f32_16x16x32_bf16 per matrix. Writes transform (tx/tc) AND the
// conv accumulators' initial value (cx/cc), since conv = h + agg.
// ---------------------------------------------------------------------------
__global__ void __launch_bounds__(512)
gemm_kernel(const float* __restrict__ x,
            const unsigned char* __restrict__ mask,
            const float* __restrict__ means,
            const float* __restrict__ var,
            const unsigned short* __restrict__ wt,
            const unsigned short* __restrict__ wsq,
            const float* __restrict__ bias,
            float* __restrict__ tx, float* __restrict__ tc,
            float* __restrict__ cx, float* __restrict__ cc) {
    __shared__ __align__(16) unsigned short As[16 * FF];   // mean_mat tile, bf16
    __shared__ __align__(16) unsigned short Av[16 * FF];   // var_mat tile,  bf16

    int tid = threadIdx.x;
    int k   = blockIdx.x / MTILES;
    int m0  = (blockIdx.x % MTILES) * 16;

    for (int i = tid; i < 16 * FF; i += 512) {
        int r = i >> 8, f = i & 255;
        int n = m0 + r;
        bool m = mask[(size_t)n * FF + f] != 0;
        float mv = m ? means[k * FF + f] : x[(size_t)n * FF + f];
        float vv = m ? var[k * FF + f] : 0.0f;
        As[i] = f2bf(mv);
        Av[i] = f2bf(vv);
    }
    __syncthreads();

    int wave = tid >> 5, lane = tid & 31;
    int n0   = wave * 16;
    int ml   = lane & 15, half = lane >> 4;

    float bv = bias[n0 + ml];
    v8f accX = {}, accC = {};

    const unsigned short* wtp = wt  + (size_t)(n0 + ml) * FF;
    const unsigned short* wsp = wsq + (size_t)(n0 + ml) * FF;

    #pragma unroll
    for (int kk = 0; kk < 8; ++kk) {
        // A 16x32 bf16 layout: lanes hold row M=lane&15; first 8 elems K=half*8..,
        // next 8 elems K=16+half*8..
        int abase = ml * FF + kk * 32 + half * 8;
        union { v4u u[2]; v16bf v; } am, av, bm, bs;
        am.u[0] = *(const v4u*)&As[abase];
        am.u[1] = *(const v4u*)&As[abase + 16];
        av.u[0] = *(const v4u*)&Av[abase];
        av.u[1] = *(const v4u*)&Av[abase + 16];
        // B 32x16 bf16 layout: lane = col n0+ml; first 8 elems K=half*8.., next 8 elems K=16+half*8..
        int bbase = kk * 32 + half * 8;
        bm.u[0] = *(const v4u*)&wtp[bbase];
        bm.u[1] = *(const v4u*)&wtp[bbase + 16];
        bs.u[0] = *(const v4u*)&wsp[bbase];
        bs.u[1] = *(const v4u*)&wsp[bbase + 16];

        accX = __builtin_amdgcn_wmma_f32_16x16x32_bf16(false, am.v, false, bm.v,
                                                       (short)0, accX, false, false);
        accC = __builtin_amdgcn_wmma_f32_16x16x32_bf16(false, av.v, false, bs.v,
                                                       (short)0, accC, false, false);
    }

    // C/D layout: VGPR i -> row M = half*8 + i, col N = lane&15
    #pragma unroll
    for (int i = 0; i < 8; ++i) {
        size_t row = (size_t)m0 + half * 8 + i;
        size_t idx = (((size_t)k * NN + row) << 8) + n0 + ml;
        float xv = accX[i] + bv;
        tx[idx] = xv;      cx[idx] = xv;
        tc[idx] = accC[i]; cc[idx] = accC[i];
    }
}

// ---------------------------------------------------------------------------
// scatter: conv agg — for each edge, c*[k,dst,:] += t*[k,src,:]
// one block per edge, 256 threads = 256 features
// ---------------------------------------------------------------------------
__global__ void scatter_kernel(const long long* __restrict__ edges,
                               const float* __restrict__ tx,
                               const float* __restrict__ tc,
                               float* __restrict__ cx,
                               float* __restrict__ cc) {
    size_t e = blockIdx.x;
    int s = (int)edges[e];
    int d = (int)edges[(size_t)EE + e];
    int f = threadIdx.x;
    #pragma unroll
    for (int k = 0; k < KC; ++k) {
        size_t si = (((size_t)k * NN + s) << 8) + f;
        size_t di = (((size_t)k * NN + d) << 8) + f;
        unsafeAtomicAdd(&cx[di], tx[si]);
        unsafeAtomicAdd(&cc[di], tc[si]);
    }
}

// ---------------------------------------------------------------------------
// combine: out[n,o] = sum_k gamma[k,n] * ex_relu(cx[k,n,o], cc[k,n,o])
// ---------------------------------------------------------------------------
__global__ void combine_kernel(const float* __restrict__ cx,
                               const float* __restrict__ cc,
                               const float* __restrict__ gamma,
                               float* __restrict__ out) {
    int n = blockIdx.x, o = threadIdx.x;
    const float inv_s2pi = 0.3989422804014327f;   // 1/sqrt(2*pi)
    const float inv_s2   = 0.7071067811865476f;   // 1/sqrt(2)
    float acc = 0.f;
    #pragma unroll
    for (int k = 0; k < KC; ++k) {
        size_t idx = (((size_t)k * NN + n) << 8) + o;
        float mu = cx[idx];
        float s2 = cc[idx];
        float g  = gamma[(size_t)k * NN + n];
        float r;
        if (s2 == 0.0f) {
            r = fmaxf(mu, 0.0f);
        } else {
            float ss = sqrtf(s2);
            float w  = mu / ss;
            r = ss * (expf(-0.5f * w * w) * inv_s2pi +
                      0.5f * w * (1.0f + erff(w * inv_s2)));
        }
        acc += g * r;
    }
    out[(size_t)n * FF + o] = acc;
}

// ---------------------------------------------------------------------------
extern "C" void kernel_launch(void* const* d_in, const int* in_sizes, int n_in,
                              void* d_out, int out_size, void* d_ws, size_t ws_size,
                              hipStream_t stream) {
    const float*         x       = (const float*)d_in[0];
    const long long*     edges   = (const long long*)d_in[1];
    const unsigned char* mask    = (const unsigned char*)d_in[2];
    const float*         logp    = (const float*)d_in[3];
    const float*         means   = (const float*)d_in[4];
    const float*         logvars = (const float*)d_in[5];
    const float*         weight  = (const float*)d_in[6];
    const float*         bias    = (const float*)d_in[7];
    float*               out     = (float*)d_out;

    char* ws = (char*)d_ws;
    unsigned short* wt  = (unsigned short*)(ws);                    // 128 KB
    unsigned short* wsq = (unsigned short*)(ws + (128u << 10));     // 128 KB
    float*          var = (float*)(ws + (256u << 10));              //   5 KB
    float*          gm  = (float*)(ws + (264u << 10));              //   1 MB

    size_t base = 2ull << 20;
    size_t tsz  = (size_t)KC * NN * FF * sizeof(float);             // ~244 MB
    float* tx = (float*)(ws + base);
    float* tc = (float*)(ws + base + tsz);
    float* cx = (float*)(ws + base + 2 * tsz);
    float* cc = (float*)(ws + base + 3 * tsz);

    prep_kernel   <<<FF + KC,      256, 0, stream>>>(weight, logvars, wt, wsq, var);
    gamma_kernel  <<<NN / 8,       256, 0, stream>>>(x, mask, logp, means, var, gm);
    gemm_kernel   <<<KC * MTILES,  512, 0, stream>>>(x, mask, means, var, wt, wsq,
                                                     bias, tx, tc, cx, cc);
    scatter_kernel<<<EE,           256, 0, stream>>>(edges, tx, tc, cx, cc);
    combine_kernel<<<NN,           256, 0, stream>>>(cx, cc, gm, out);
}
